// PyWTWaveletUpsample_76347338653734
// MI455X (gfx1250) — matrix operand order repaired and checked by reading, not compiled, not run on hardware
//
#include <hip/hip_runtime.h>
#include <hip/hip_fp16.h>

typedef __attribute__((ext_vector_type(16))) _Float16 v16h;
typedef __attribute__((ext_vector_type(8)))  float    v8f;

union Frag16 { v16h h; unsigned int u[8]; };

#define B_    16
#define CIN_  64
#define H_    128
#define W_    128
#define H2_   256
#define W2_   256
#define K1_   576      // 9 taps * 64 ch
#define NCHUNK 18      // K1_/32

// ---------------------------------------------------------------------------
// Pack OIHW f32 weights into WMMA B-fragment order (f16 pairs, one dword per
// lane-VGPR).  Output layout: [o_tile][chunk][lane][v] dwords -> per lane the
// 8 dwords of a chunk are contiguous (b128) and coalesced across lanes.
// K ordering: k = (kh*3+kw)*64 + ci.  B layout (wave32, 16-bit, 32x16):
//   lane l holds column n=l&15; k = (l>=16 ? 8 : 0) + 2v + h + (v>=4 ? 8 : 0)  within chunk.
// ---------------------------------------------------------------------------
__global__ void pack_weights_kernel(const float* __restrict__ w,
                                    unsigned int* __restrict__ wp,
                                    int total) {
    int gid = blockIdx.x * blockDim.x + threadIdx.x;
    if (gid >= total) return;
    int v  = gid & 7;
    int l  = (gid >> 3) & 31;
    int q  = (gid >> 8) % NCHUNK;
    int ot = gid / (NCHUNK * 256);
    int o  = ot * 16 + (l & 15);
    int kl0 = ((l >> 4) << 3) + 2 * v + ((v >> 2) << 3);
    unsigned int packed = 0;
    #pragma unroll
    for (int h = 0; h < 2; ++h) {
        int k   = q * 32 + kl0 + h;       // global K index
        int tap = k >> 6;                 // kh*3+kw
        int ci  = k & 63;
        float val = w[(o * 64 + ci) * 9 + tap];
        _Float16 hv = (_Float16)val;
        unsigned short bits;
        __builtin_memcpy(&bits, &hv, 2);
        packed |= ((unsigned int)bits) << (16 * h);
    }
    wp[gid] = packed;
}

// ---------------------------------------------------------------------------
// Kernel 1: conv1 (implicit GEMM, f16 WMMA) fused with wavelet combine and
// 2x upsample.  Block = 128 threads = 4 waves, tile = 32 pixels of one row.
// Wave nt owns out-channels [nt*16, nt*16+16) for all 4 wavelet groups and
// both 16-pixel sub-tiles: 8 accumulators, each B fragment reused 2x.
// Output `up` (NHWC f16) written as two contiguous 8KB rows per block via an
// LDS transpose (aliases the input staging buffer).
// ---------------------------------------------------------------------------
__global__ void __launch_bounds__(128)
conv1_wavelet_up_kernel(const float* __restrict__ x,
                        const unsigned int* __restrict__ wpack,
                        const float* __restrict__ b_pre,
                        _Float16* __restrict__ up) {
    // union: input tile xs[3][34][64] (13056 B) / output patch os[2][64][64] (16384 B)
    __shared__ __align__(16) _Float16 smem[2 * 64 * 64];
    _Float16* xs = smem;
    _Float16* os = smem;

    int bi = blockIdx.x;
    int wt = bi & 3;                // 4 tiles of 32 per row of 128
    int h  = (bi >> 2) & 127;
    int b  = bi >> 9;
    int w0 = wt * 32;
    int tid = threadIdx.x;

    // Stage x tile (rows h-1..h+1, cols w0-1..w0+32, all 64 ch) into LDS f16.
    for (int idx = tid; idx < 3 * 34 * 64; idx += 128) {
        int col = idx % 34;
        int ci  = (idx / 34) & 63;
        int row = idx / (34 * 64);
        int yy = h + row - 1;
        int xx = w0 + col - 1;
        float val = 0.0f;
        if (yy >= 0 && yy < H_ && xx >= 0 && xx < W_)
            val = x[((size_t)(b * 64 + ci) * H_ + yy) * W_ + xx];
        xs[(row * 34 + col) * 64 + ci] = (_Float16)val;
    }
    __syncthreads();

    int lane = tid & 31;
    int nt   = tid >> 5;            // wave id = base-channel tile
    int m    = lane & 15;           // A-matrix row (pixel within sub-tile)
    int kgrp = lane >> 4;           // which K-octet this lane holds

    v8f acc[4][2];
    #pragma unroll
    for (int g = 0; g < 4; ++g)
        #pragma unroll
        for (int t = 0; t < 2; ++t)
            #pragma unroll
            for (int i = 0; i < 8; ++i) acc[g][t][i] = 0.0f;

    for (int q = 0; q < NCHUNK; ++q) {
        int tap = q >> 1;
        int dy = tap / 3, dx = tap % 3;
        int cibase = (q & 1) * 32;
        Frag16 a[2];
        #pragma unroll
        for (int t = 0; t < 2; ++t) {
            const unsigned int* xp =
                (const unsigned int*)&xs[(dy * 34 + m + dx + t * 16) * 64 + cibase];
            #pragma unroll
            for (int v = 0; v < 4; ++v) {
                a[t].u[v]     = xp[kgrp * 4 + v];       // K = kgrp*8 + 2v (+1)
                a[t].u[v + 4] = xp[8 + kgrp * 4 + v];   // K = 16 + kgrp*8 + 2v (+1)
            }
        }
        #pragma unroll
        for (int g = 0; g < 4; ++g) {
            int ot = g * 4 + nt;                        // out-channel tile of 256
            const unsigned int* bp =
                &wpack[((size_t)(ot * NCHUNK + q)) * 256 + lane * 8];
            Frag16 bf;
            #pragma unroll
            for (int v = 0; v < 8; ++v) bf.u[v] = bp[v];
            acc[g][0] = __builtin_amdgcn_wmma_f32_16x16x32_f16(
                false, a[0].h, false, bf.h, (short)0, acc[g][0], false, false);
            acc[g][1] = __builtin_amdgcn_wmma_f32_16x16x32_f16(
                false, a[1].h, false, bf.h, (short)0, acc[g][1], false, false);
        }
    }

    __syncthreads();   // all waves done reading xs; safe to alias as os

    // Bias + wavelet combine; write into os[py][xx(64)][c(64)].
    int n = lane & 15;
    int c = nt * 16 + n;
    float bll = b_pre[c];
    float blh = b_pre[64 + c];
    float bhl = b_pre[128 + c];
    float bhh = b_pre[192 + c];
    #pragma unroll
    for (int t = 0; t < 2; ++t) {
        #pragma unroll
        for (int v = 0; v < 8; ++v) {
            int p = t * 16 + v + kgrp * 8;              // pixel within 32-tile
            float ll = acc[0][t][v] + bll;
            float lh = acc[1][t][v] + blh;
            float hl = acc[2][t][v] + bhl;
            float hh = acc[3][t][v] + bhh;
            float ee = 0.5f * (ll + lh + hl + hh);
            float eo = 0.5f * (ll - lh + hl - hh);
            float oe = 0.5f * (ll + lh - hl - hh);
            float oo = 0.5f * (ll - lh - hl + hh);
            int xx = 2 * p;
            os[(0 * 64 + xx) * 64 + c]       = (_Float16)ee;
            os[(0 * 64 + xx + 1) * 64 + c]   = (_Float16)eo;
            os[(1 * 64 + xx) * 64 + c]       = (_Float16)oe;
            os[(1 * 64 + xx + 1) * 64 + c]   = (_Float16)oo;
        }
    }
    __syncthreads();

    // Coalesced store: two contiguous 8KB rows (64 px * 64 ch f16 each).
    {
        const uint4* osv = (const uint4*)os;            // 1024 uint4 chunks
        #pragma unroll
        for (int j = 0; j < 8; ++j) {
            int chunk = j * 128 + tid;                  // 0..1023
            int py  = chunk >> 9;                       // 512 uint4 per row
            int rem = chunk & 511;
            size_t goff = (((size_t)(b * H2_ + 2 * h + py)) * W2_ + 2 * w0) * 64;
            ((uint4*)(up + goff))[rem] = osv[chunk];
        }
    }
}

// ---------------------------------------------------------------------------
// Kernel 2: conv2 over `up` (NHWC f16) -> out (NCHW f32).  Block = 4 waves,
// tile = 32 pixels of one row; each wave: 2 accumulators (B reused 2x).
// Output transposed through LDS for contiguous 64B stores per thread.
// ---------------------------------------------------------------------------
__global__ void __launch_bounds__(128)
conv2_kernel(const _Float16* __restrict__ up,
             const unsigned int* __restrict__ wpack,
             const float* __restrict__ b_post,
             float* __restrict__ out) {
    __shared__ __align__(16) _Float16 xs[3 * 34 * 64];  // 13056 B
    __shared__ __align__(16) float outs[64 * 32];       // 8192 B

    int bi = blockIdx.x;
    int xt = bi & 7;                // 8 tiles of 32 per row of 256
    int y  = (bi >> 3) & 255;
    int b  = bi >> 11;
    int x0 = xt * 32;
    int tid = threadIdx.x;

    // Stage up tile: rows y-1..y+1, cols x0-1..x0+32, 64 ch (dword copies).
    for (int idx = tid; idx < 3 * 34 * 32; idx += 128) {
        int cu  = idx & 31;
        int col = (idx >> 5) % 34;
        int row = idx / (34 * 32);
        int yy = y + row - 1;
        int xx = x0 + col - 1;
        unsigned int val = 0;
        if (yy >= 0 && yy < H2_ && xx >= 0 && xx < W2_)
            val = ((const unsigned int*)up)[(((size_t)(b * H2_ + yy)) * W2_ + xx) * 32 + cu];
        ((unsigned int*)xs)[(row * 34 + col) * 32 + cu] = val;
    }
    __syncthreads();

    int lane = tid & 31;
    int nt   = tid >> 5;
    int m    = lane & 15;
    int kgrp = lane >> 4;

    v8f acc[2];
    #pragma unroll
    for (int t = 0; t < 2; ++t)
        #pragma unroll
        for (int i = 0; i < 8; ++i) acc[t][i] = 0.0f;

    for (int q = 0; q < NCHUNK; ++q) {
        int tap = q >> 1;
        int dy = tap / 3, dx = tap % 3;
        int cibase = (q & 1) * 32;
        Frag16 a[2];
        #pragma unroll
        for (int t = 0; t < 2; ++t) {
            const unsigned int* xp =
                (const unsigned int*)&xs[(dy * 34 + m + dx + t * 16) * 64 + cibase];
            #pragma unroll
            for (int v = 0; v < 4; ++v) {
                a[t].u[v]     = xp[kgrp * 4 + v];
                a[t].u[v + 4] = xp[8 + kgrp * 4 + v];
            }
        }
        const unsigned int* bp =
            &wpack[((size_t)(nt * NCHUNK + q)) * 256 + lane * 8];
        Frag16 bf;
        #pragma unroll
        for (int v = 0; v < 8; ++v) bf.u[v] = bp[v];
        acc[0] = __builtin_amdgcn_wmma_f32_16x16x32_f16(
            false, a[0].h, false, bf.h, (short)0, acc[0], false, false);
        acc[1] = __builtin_amdgcn_wmma_f32_16x16x32_f16(
            false, a[1].h, false, bf.h, (short)0, acc[1], false, false);
    }

    // Bias, transpose through LDS, coalesced NCHW store.
    int n = lane & 15;
    int c = nt * 16 + n;
    float bias = b_post[c];
    #pragma unroll
    for (int t = 0; t < 2; ++t)
        #pragma unroll
        for (int v = 0; v < 8; ++v) {
            int p = t * 16 + v + kgrp * 8;
            outs[c * 32 + p] = acc[t][v] + bias;
        }
    __syncthreads();
    {
        int c2 = tid >> 1;                 // 0..63
        int xh = (tid & 1) * 16;           // half-row of 16 floats
        size_t obase = (((size_t)(b * 64 + c2)) * H2_ + y) * W2_ + x0 + xh;
        const float4* ov = (const float4*)&outs[c2 * 32 + xh];
        #pragma unroll
        for (int i = 0; i < 4; ++i)
            ((float4*)(out + obase))[i] = ov[i];
    }
}

// ---------------------------------------------------------------------------
extern "C" void kernel_launch(void* const* d_in, const int* in_sizes, int n_in,
                              void* d_out, int out_size, void* d_ws, size_t ws_size,
                              hipStream_t stream) {
    const float* x      = (const float*)d_in[0];
    const float* w_pre  = (const float*)d_in[1];
    const float* b_pre  = (const float*)d_in[2];
    const float* w_post = (const float*)d_in[3];
    const float* b_post = (const float*)d_in[4];
    float* out = (float*)d_out;

    unsigned char* ws = (unsigned char*)d_ws;
    // workspace layout: packed w_pre (288 KB) | packed w_post (72 KB) | up (128 MB f16)
    unsigned int* wp_pre  = (unsigned int*)ws;
    unsigned int* wp_post = (unsigned int*)(ws + 16 * NCHUNK * 256 * 4);
    _Float16*     up      = (_Float16*)(ws + 16 * NCHUNK * 256 * 4 + 4 * NCHUNK * 256 * 4);

    int tot_pre  = 16 * NCHUNK * 256;   // 73728 dwords
    int tot_post = 4 * NCHUNK * 256;    // 18432 dwords
    pack_weights_kernel<<<(tot_pre  + 255) / 256, 256, 0, stream>>>(w_pre,  wp_pre,  tot_pre);
    pack_weights_kernel<<<(tot_post + 255) / 256, 256, 0, stream>>>(w_post, wp_post, tot_post);

    conv1_wavelet_up_kernel<<<B_ * H_ * (W_ / 32), 128, 0, stream>>>(x, wp_pre, b_pre, up);
    conv2_kernel<<<B_ * H2_ * (W2_ / 32), 128, 0, stream>>>(up, wp_post, b_post, out);
}
